// SimpleAttention_24739011625862
// MI455X (gfx1250) — matrix operand order repaired and checked by reading, not compiled, not run on hardware
//
#include <hip/hip_runtime.h>

// ---------------------------------------------------------------------------
// CDNA5 (gfx1250) bf16-WMMA causal self-attention forward.
// GEMMs: v_wmma_f32_16x16x32_bf16 + global_load_async_to_lds_b128 double
// buffering (ASYNCcnt).  Attention: 32 q-rows/wave flash softmax, WMMA both ways.
// ---------------------------------------------------------------------------

typedef __attribute__((ext_vector_type(16))) __bf16       v16bf;
typedef __attribute__((ext_vector_type(8)))  float        v8f;
typedef __attribute__((ext_vector_type(4)))  unsigned int v4u;

union FragU { v16bf v; v4u q[2]; };

#define N_EMBD 1024
#define N_HEAD 16
#define HEADD  64
#define SEQ    2048
#define BATCH  2
#define BT     (BATCH * SEQ)      // 4096 tokens

// fp32 -> bf16 round-to-nearest-even (bit manipulation; no __bf16 arithmetic)
__device__ __forceinline__ unsigned short f2bf(float x) {
  unsigned int u = __float_as_uint(x);
  u += 0x7FFFu + ((u >> 16) & 1u);
  return (unsigned short)(u >> 16);
}

// A-fragment (16xK slice, row-major source, 16-bit elements):
// lane l: M = l&15; low lanes hold K {k0..k0+7, k0+16..k0+23},
// high lanes hold K {k0+8..k0+15, k0+24..k0+31}.  Two 16B loads.
__device__ __forceinline__ v16bf load_frag_a(const unsigned short* base, int ld, int k0) {
  const int lane = threadIdx.x & 31;
  const unsigned short* p = base + (size_t)(lane & 15) * ld + k0 + ((lane >> 4) << 3);
  FragU f;
  f.q[0] = *(const v4u*)(p);
  f.q[1] = *(const v4u*)(p + 16);
  return f.v;
}

// B-fragment (Kx16 slice) from an [N][K] row-major array:
// lane l: N = l&15; low lanes hold K {k0..k0+7, k0+16..k0+23},
// high lanes hold K {k0+8..k0+15, k0+24..k0+31}.
__device__ __forceinline__ v16bf load_frag_b(const unsigned short* base, int ld, int k0) {
  const int lane = threadIdx.x & 31;
  const unsigned short* p = base + (size_t)(lane & 15) * ld + k0 + ((lane >> 4) << 3);
  FragU f;
  f.q[0] = *(const v4u*)(p);
  f.q[1] = *(const v4u*)(p + 16);
  return f.v;
}

__device__ __forceinline__ v8f wmma_bf16(v16bf a, v16bf b, v8f c) {
  return __builtin_amdgcn_wmma_f32_16x16x32_bf16(false, a, false, b, (short)0, c,
                                                 false, false);
}

// Async copy 16B global -> LDS, tracked by ASYNCcnt (CDNA5 GV form).
__device__ __forceinline__ void async_ld_b128(unsigned lds_off, const unsigned short* g) {
  asm volatile("global_load_async_to_lds_b128 %0, %1, off"
               :: "v"(lds_off), "v"(g) : "memory");
}
__device__ __forceinline__ void wait_async_le4() {
  asm volatile("s_wait_asynccnt 0x4" ::: "memory");
}
__device__ __forceinline__ void wait_async_0() {
  asm volatile("s_wait_asynccnt 0x0" ::: "memory");
}
__device__ __forceinline__ void wait_ds_0() {
  asm volatile("s_wait_dscnt 0x0" ::: "memory");
}
// generic pointer to LDS: low 32 bits are the LDS byte offset
__device__ __forceinline__ unsigned lds_off(const void* p) {
  return (unsigned)(unsigned long long)p;
}

// ---------------------------------------------------------------------------
// 1) fp32 -> bf16 elementwise
__global__ __launch_bounds__(256) void k_cvt(const float* __restrict__ in,
                                             unsigned short* __restrict__ out, int n) {
  int i = blockIdx.x * 256 + threadIdx.x;
  if (i < n) out[i] = f2bf(in[i]);
}

// 2) fp32 [1024][N] -> bf16 [N][1024] (transpose; K hardcoded -> shifts only)
__global__ __launch_bounds__(256) void k_cvt_T(const float* __restrict__ w,
                                               unsigned short* __restrict__ wT, int N) {
  int i = blockIdx.x * 256 + threadIdx.x;
  if (i < (N << 10)) {
    const int n = i >> 10, k = i & 1023;
    wT[i] = f2bf(w[(size_t)k * N + n]);
  }
}

// ---------------------------------------------------------------------------
// 3) QKV GEMM: [4096 x 1024] x [1024 x 3072] + bias.
//    One wave -> 32(M) x 64(N) tile, 8 WMMAs per K=32 step.
//    A tile (32x32, 2KB) double-buffered in LDS via async-to-LDS copies.
__global__ __launch_bounds__(32) void k_qkv(const unsigned short* __restrict__ xb,
                                            const unsigned short* __restrict__ waT,
                                            const float* __restrict__ bias,
                                            unsigned short* __restrict__ qws,
                                            unsigned short* __restrict__ kws,
                                            unsigned short* __restrict__ vtws) {
  __shared__ __align__(16) unsigned short Abuf[2][32 * 32];
  const int mt = blockIdx.y;      // 0..127 : 32 tokens
  const int nt = blockIdx.x;      // 0..47  : 64 output cols
  const int lane = threadIdx.x & 31;
  const int hi = lane >> 4, nl = lane & 15;

  const unsigned short* abase = xb + (size_t)mt * 32 * N_EMBD;

  auto issueA = [&](int buf, int k0) {
#pragma unroll
    for (int i = 0; i < 4; ++i) {
      const int c = i * 32 + lane;            // 128 chunks x 16B = 32x32 bf16
      const int row = c >> 2, cb = c & 3;
      async_ld_b128(lds_off(&Abuf[buf][0]) + c * 16,
                    abase + (size_t)row * N_EMBD + k0 + cb * 8);
    }
  };

  issueA(0, 0);
  v8f acc[8] = {};
  int cur = 0;
#pragma unroll 2
  for (int k0 = 0; k0 < N_EMBD; k0 += 32) {
    if (k0 + 32 < N_EMBD) {
      wait_ds_0();                 // prior frag reads of buf cur^1 complete
      issueA(cur ^ 1, k0 + 32);
      __builtin_prefetch(waT + (size_t)(nt * 64 + nl) * N_EMBD + k0 + 32, 0, 3);
      wait_async_le4();            // current buffer's 4 copies have landed
    } else {
      wait_async_0();
    }
    const v16bf a0 = load_frag_a(&Abuf[cur][0],       32, 0);   // rows 0..15
    const v16bf a1 = load_frag_a(&Abuf[cur][16 * 32], 32, 0);   // rows 16..31
#pragma unroll
    for (int t = 0; t < 4; ++t) {
      v16bf b = load_frag_b(waT + (size_t)(nt * 64 + 16 * t) * N_EMBD, N_EMBD, k0);
      acc[t]     = wmma_bf16(a0, b, acc[t]);
      acc[4 + t] = wmma_bf16(a1, b, acc[4 + t]);
    }
    cur ^= 1;
  }

#pragma unroll
  for (int t = 0; t < 4; ++t) {
    const int col = nt * 64 + 16 * t + nl;   // 0..3071
    const float bv = bias[col];
    const int sec = col >> 10;               // 0=q 1=k 2=v
    const int h = (col & 1023) >> 6;
    const int d = col & 63;
#pragma unroll
    for (int mi = 0; mi < 2; ++mi) {
#pragma unroll
      for (int j = 0; j < 8; ++j) {
        const int token = mt * 32 + mi * 16 + j + hi * 8;   // C-layout: M=j+8*hi
        const int bb = token >> 11, tl = token & (SEQ - 1);
        const unsigned short val = f2bf(acc[mi * 4 + t][j] + bv);
        const size_t bh = (size_t)(bb * N_HEAD + h);
        if (sec == 0)      qws[(bh * SEQ + tl) * HEADD + d] = val;
        else if (sec == 1) kws[(bh * SEQ + tl) * HEADD + d] = val;
        else               vtws[(bh * HEADD + d) * SEQ + tl] = val;   // transposed
      }
    }
  }
}

// ---------------------------------------------------------------------------
// Online-softmax update for one 16-row score tile pair (cols kv..kv+31).
// s0/s1 in C-layout (element j = row row0+j+8*hi, col kv+nl / kv+16+nl).
// Rescales oacc[0..3] in place, writes P (bf16) to plds[16][32].
__device__ __forceinline__ void softmax_update(v8f s0, v8f s1, int row0, int kv,
                                               float* mrow, float* lrow, v8f* oacc,
                                               unsigned short* plds) {
  const int lane = threadIdx.x & 31;
  const int hi = lane >> 4, nl = lane & 15;
  const float sc = 0.125f;   // 1/sqrt(64)
#pragma unroll
  for (int j = 0; j < 8; ++j) {
    const int row = row0 + j + hi * 8;
    float v0 = (kv + nl      <= row) ? s0[j] * sc : -1e30f;
    float v1 = (kv + 16 + nl <= row) ? s1[j] * sc : -1e30f;
    float t = fmaxf(v0, v1);
#pragma unroll
    for (int m = 1; m < 16; m <<= 1) t = fmaxf(t, __shfl_xor(t, m, 32));
    const float mnew = fmaxf(mrow[j], t);
    const float a  = __expf(mrow[j] - mnew);
    const float p0 = __expf(v0 - mnew);
    const float p1 = __expf(v1 - mnew);
    float rs = p0 + p1;
#pragma unroll
    for (int m = 1; m < 16; m <<= 1) rs += __shfl_xor(rs, m, 32);
    lrow[j] = lrow[j] * a + rs;
    mrow[j] = mnew;
#pragma unroll
    for (int t4 = 0; t4 < 4; ++t4) oacc[t4][j] = oacc[t4][j] * a;
    plds[(j + hi * 8) * 32 + nl]      = f2bf(p0);
    plds[(j + hi * 8) * 32 + nl + 16] = f2bf(p1);
  }
}

// 4) Flash-style causal attention: one wave per (b, h, 32-row q tile).
//    kv loop step 32: 8 WMMAs for S (two q sub-tiles share K-fragments),
//    online softmax, 8 WMMAs for P V (shared V-fragments).
__global__ __launch_bounds__(32) void k_attn(const unsigned short* __restrict__ qws,
                                             const unsigned short* __restrict__ kws,
                                             const unsigned short* __restrict__ vtws,
                                             unsigned short* __restrict__ attnO) {
  __shared__ __align__(16) unsigned short Plds[2][16 * 32];
  const int bid = blockIdx.x;
  const int qt = bid & 63;           // 64 q-tiles of 32 rows
  const int h  = (bid >> 6) & 15;
  const int bb = bid >> 10;
  const int q0 = qt * 32;
  const int lane = threadIdx.x & 31;
  const int hi = lane >> 4, nl = lane & 15;

  const size_t bh = (size_t)(bb * N_HEAD + h);
  const unsigned short* qp = qws + bh * SEQ * HEADD;
  const unsigned short* kp = kws + bh * SEQ * HEADD;
  const unsigned short* vp = vtws + bh * HEADD * SEQ;

  const v16bf qa0 = load_frag_a(qp + (size_t)q0 * HEADD,        HEADD, 0);
  const v16bf qa1 = load_frag_a(qp + (size_t)q0 * HEADD,        HEADD, 32);
  const v16bf qb0 = load_frag_a(qp + (size_t)(q0 + 16) * HEADD, HEADD, 0);
  const v16bf qb1 = load_frag_a(qp + (size_t)(q0 + 16) * HEADD, HEADD, 32);

  float mA[8], lA[8], mB[8], lB[8];
#pragma unroll
  for (int j = 0; j < 8; ++j) { mA[j] = mB[j] = -1e30f; lA[j] = lB[j] = 0.f; }
  v8f oA[4] = {}, oB[4] = {};

  for (int kv = 0; kv < q0 + 32; kv += 32) {
    const v16bf kb00 = load_frag_b(kp + (size_t)kv * HEADD,        HEADD, 0);
    const v16bf kb01 = load_frag_b(kp + (size_t)kv * HEADD,        HEADD, 32);
    const v16bf kb10 = load_frag_b(kp + (size_t)(kv + 16) * HEADD, HEADD, 0);
    const v16bf kb11 = load_frag_b(kp + (size_t)(kv + 16) * HEADD, HEADD, 32);

    v8f sA0 = {}, sA1 = {}, sB0 = {}, sB1 = {};
    sA0 = wmma_bf16(qa0, kb00, sA0);  sA0 = wmma_bf16(qa1, kb01, sA0);
    sA1 = wmma_bf16(qa0, kb10, sA1);  sA1 = wmma_bf16(qa1, kb11, sA1);
    sB0 = wmma_bf16(qb0, kb00, sB0);  sB0 = wmma_bf16(qb1, kb01, sB0);
    sB1 = wmma_bf16(qb0, kb10, sB1);  sB1 = wmma_bf16(qb1, kb11, sB1);

    softmax_update(sA0, sA1, q0,      kv, mA, lA, oA, &Plds[0][0]);
    softmax_update(sB0, sB1, q0 + 16, kv, mB, lB, oB, &Plds[1][0]);

    __syncthreads();                       // single wave: dscnt ordering
    const v16bf paA = load_frag_a(&Plds[0][0], 32, 0);
    const v16bf paB = load_frag_a(&Plds[1][0], 32, 0);
    __syncthreads();

#pragma unroll
    for (int t4 = 0; t4 < 4; ++t4) {
      const v16bf vb = load_frag_b(vp + (size_t)(16 * t4) * SEQ, SEQ, kv);  // V^T
      oA[t4] = wmma_bf16(paA, vb, oA[t4]);
      oB[t4] = wmma_bf16(paB, vb, oB[t4]);
    }
  }

#pragma unroll
  for (int t4 = 0; t4 < 4; ++t4) {
#pragma unroll
    for (int j = 0; j < 8; ++j) {
      const int ta = bb * SEQ + q0 + j + hi * 8;
      attnO[(size_t)ta * N_EMBD + h * HEADD + 16 * t4 + nl] = f2bf(oA[t4][j] / lA[j]);
      const int tb = ta + 16;
      attnO[(size_t)tb * N_EMBD + h * HEADD + 16 * t4 + nl] = f2bf(oB[t4][j] / lB[j]);
    }
  }
}

// ---------------------------------------------------------------------------
// 5) Output projection: [4096 x 1024] x [1024 x 1024] + bias -> fp32 d_out
//    Same 32x64 tile + async-LDS A staging as the QKV GEMM.
__global__ __launch_bounds__(32) void k_proj(const unsigned short* __restrict__ ab,
                                             const unsigned short* __restrict__ wpT,
                                             const float* __restrict__ bias,
                                             float* __restrict__ out) {
  __shared__ __align__(16) unsigned short Abuf[2][32 * 32];
  const int mt = blockIdx.y, nt = blockIdx.x;
  const int lane = threadIdx.x & 31, hi = lane >> 4, nl = lane & 15;
  const unsigned short* abase = ab + (size_t)mt * 32 * N_EMBD;

  auto issueA = [&](int buf, int k0) {
#pragma unroll
    for (int i = 0; i < 4; ++i) {
      const int c = i * 32 + lane;
      const int row = c >> 2, cb = c & 3;
      async_ld_b128(lds_off(&Abuf[buf][0]) + c * 16,
                    abase + (size_t)row * N_EMBD + k0 + cb * 8);
    }
  };

  issueA(0, 0);
  v8f acc[8] = {};
  int cur = 0;
#pragma unroll 2
  for (int k0 = 0; k0 < N_EMBD; k0 += 32) {
    if (k0 + 32 < N_EMBD) {
      wait_ds_0();
      issueA(cur ^ 1, k0 + 32);
      __builtin_prefetch(wpT + (size_t)(nt * 64 + nl) * N_EMBD + k0 + 32, 0, 3);
      wait_async_le4();
    } else {
      wait_async_0();
    }
    const v16bf a0 = load_frag_a(&Abuf[cur][0],       32, 0);
    const v16bf a1 = load_frag_a(&Abuf[cur][16 * 32], 32, 0);
#pragma unroll
    for (int t = 0; t < 4; ++t) {
      v16bf b = load_frag_b(wpT + (size_t)(nt * 64 + 16 * t) * N_EMBD, N_EMBD, k0);
      acc[t]     = wmma_bf16(a0, b, acc[t]);
      acc[4 + t] = wmma_bf16(a1, b, acc[4 + t]);
    }
    cur ^= 1;
  }

#pragma unroll
  for (int t = 0; t < 4; ++t) {
    const int col = nt * 64 + 16 * t + nl;
    const float bv = bias[col];
#pragma unroll
    for (int mi = 0; mi < 2; ++mi) {
#pragma unroll
      for (int j = 0; j < 8; ++j) {
        const int row = mt * 32 + mi * 16 + j + hi * 8;
        out[(size_t)row * N_EMBD + col] = acc[mi * 4 + t][j] + bv;
      }
    }
  }
}

// ---------------------------------------------------------------------------
extern "C" void kernel_launch(void* const* d_in, const int* in_sizes, int n_in,
                              void* d_out, int out_size, void* d_ws, size_t ws_size,
                              hipStream_t stream) {
  (void)in_sizes; (void)n_in; (void)out_size; (void)ws_size;
  const float* x      = (const float*)d_in[0];
  const float* w_attn = (const float*)d_in[1];
  const float* b_attn = (const float*)d_in[2];
  const float* w_proj = (const float*)d_in[3];
  const float* b_proj = (const float*)d_in[4];
  float* out = (float*)d_out;

  // workspace layout (bf16 buffers), ~48 MB total
  char* w = (char*)d_ws;
  unsigned short* xb   = (unsigned short*)(w);                        //  8 MB x bf16
  unsigned short* waT  = (unsigned short*)(w + (8ull  << 20));        //  6 MB w_attn^T
  unsigned short* wpT  = (unsigned short*)(w + (14ull << 20));        //  2 MB w_proj^T
  unsigned short* qws  = (unsigned short*)(w + (16ull << 20));        //  8 MB q [B][H][T][D]
  unsigned short* kws  = (unsigned short*)(w + (24ull << 20));        //  8 MB k [B][H][T][D]
  unsigned short* vtws = (unsigned short*)(w + (32ull << 20));        //  8 MB v [B][H][D][T]
  unsigned short* aO   = (unsigned short*)(w + (40ull << 20));        //  8 MB attn out bf16

  k_cvt  <<<(BT * N_EMBD + 255) / 256, 256, 0, stream>>>(x, xb, BT * N_EMBD);
  k_cvt_T<<<(3 * N_EMBD * N_EMBD + 255) / 256, 256, 0, stream>>>(w_attn, waT, 3 * N_EMBD);
  k_cvt_T<<<(N_EMBD * N_EMBD + 255) / 256, 256, 0, stream>>>(w_proj, wpT, N_EMBD);
  k_qkv  <<<dim3(48, 128), 32, 0, stream>>>(xb, waT, b_attn, qws, kws, vtws);
  k_attn <<<dim3(BATCH * N_HEAD * (SEQ / 32)), 32, 0, stream>>>(qws, kws, vtws, aO);
  k_proj <<<dim3(16, 128), 32, 0, stream>>>(aO, wpT, b_proj, out);
}
